// STFTKANLayer_66340064854259
// MI455X (gfx1250) — matrix operand order repaired and checked by reading, not compiled, not run on hardware
//
#include <hip/hip_runtime.h>

// ---------------------------------------------------------------------------
// STFT-KAN layer collapsed to y = x @ W^T + bias  (W precomputed each call)
//   x  : [4096, 2048] f32      W : [1024, 2048] f32 -> bf16 hi/lo split
//   y  : [4096, 1024] f32
// GEMM done with v_wmma_f32_16x16x32_bf16 using the bf16x3 decomposition
// (xh*wh + xh*wl + xl*wh), fp32 accumulation.
// Wave tile 64x32 (4x2 WMMA tiles) -> 24 b128 loads : 24 WMMAs per 32-K step.
// ---------------------------------------------------------------------------

#define INPUTDIM 2048
#define OUTDIM   1024
#define GRIDSZ   64
#define WINSZ    256
#define STRIDE_  128
#define NUM_WIN  15
#define BATCH    4096

typedef __attribute__((ext_vector_type(16))) __bf16         v16bf;
typedef __attribute__((ext_vector_type(8)))  float          v8f;
typedef __attribute__((ext_vector_type(4)))  unsigned int   u32x4;
typedef __attribute__((ext_vector_type(4)))  unsigned short u16x4;
typedef __attribute__((ext_vector_type(4)))  float          f32x4;

union FragU { v16bf v; u32x4 q[2]; };

__device__ __forceinline__ unsigned short f2bf_rne(float f) {
    unsigned u = __float_as_uint(f);
    unsigned r = u + 0x7FFFu + ((u >> 16) & 1u);   // round-to-nearest-even
    return (unsigned short)(r >> 16);
}
__device__ __forceinline__ float bf2f(unsigned short h) {
    return __uint_as_float(((unsigned)h) << 16);
}

// --------------------------- kernel 1: trig tables -------------------------
// cwT[g][j] = win[j]*cos(2*pi*j*(g+1)/256),  swT likewise (transposed g-major
// so that consecutive-j lanes in the W-build kernel read coalesced rows).
__global__ __launch_bounds__(256) void k_tables(float* __restrict__ cwT,
                                                float* __restrict__ swT) {
    int t = blockIdx.x * 256 + threadIdx.x;      // 64*256 = 16384 threads
    int g = t >> 8;
    int j = t & 255;
    const float twopi = 6.28318530717958647692f;
    float jn = (float)j;
    float kf = (float)(g + 1);
    // hann (periodic) normalized: sum over 256 samples is exactly 128
    float win = 0.5f * (1.0f - cosf(twopi * jn / 256.0f)) * (1.0f / 128.0f);
    float ang = twopi * jn * kf / 256.0f;
    float s, c;
    sincosf(ang, &s, &c);
    cwT[g * 256 + j] = win * c;
    swT[g * 256 + j] = win * s;
}

// --------------------------- kernel 2: build W -----------------------------
// One thread per (o,i).  At most two windows cover sample i.
__global__ __launch_bounds__(256) void k_buildw(const float* __restrict__ fc,
                                                const float* __restrict__ cwT,
                                                const float* __restrict__ swT,
                                                unsigned short* __restrict__ wh,
                                                unsigned short* __restrict__ wl) {
    int t = blockIdx.x * 256 + threadIdx.x;      // OUTDIM*INPUTDIM threads
    int i = t & (INPUTDIM - 1);
    int o = t >> 11;
    const float* fc0 = fc;
    const float* fc1 = fc + OUTDIM * NUM_WIN * GRIDSZ;

    float acc = 0.0f;
    int n1 = i >> 7;                              // window with j in [0,128)
    if (n1 < NUM_WIN) {
        int j = i - n1 * STRIDE_;
        const float* f0 = fc0 + (o * NUM_WIN + n1) * GRIDSZ;
        const float* f1 = fc1 + (o * NUM_WIN + n1) * GRIDSZ;
        #pragma unroll 8
        for (int g = 0; g < GRIDSZ; ++g)
            acc += cwT[g * 256 + j] * f0[g] + swT[g * 256 + j] * f1[g];
    }
    int n0 = n1 - 1;                              // window with j in [128,256)
    if (n0 >= 0) {
        int j = i - n0 * STRIDE_;
        const float* f0 = fc0 + (o * NUM_WIN + n0) * GRIDSZ;
        const float* f1 = fc1 + (o * NUM_WIN + n0) * GRIDSZ;
        #pragma unroll 8
        for (int g = 0; g < GRIDSZ; ++g)
            acc += cwT[g * 256 + j] * f0[g] + swT[g * 256 + j] * f1[g];
    }
    unsigned short h = f2bf_rne(acc);
    unsigned short l = f2bf_rne(acc - bf2f(h));
    wh[t] = h;
    wl[t] = l;
}

// --------------------------- kernel 3: split x -----------------------------
__global__ __launch_bounds__(256) void k_splitx(const float* __restrict__ x,
                                                unsigned short* __restrict__ xh,
                                                unsigned short* __restrict__ xl) {
    int t = blockIdx.x * 256 + threadIdx.x;      // (BATCH*INPUTDIM)/4 threads
    f32x4 v = ((const f32x4*)x)[t];
    u16x4 h, l;
    #pragma unroll
    for (int e = 0; e < 4; ++e) {
        unsigned short hh = f2bf_rne(v[e]);
        h[e] = hh;
        l[e] = f2bf_rne(v[e] - bf2f(hh));
    }
    ((u16x4*)xh)[t] = h;
    ((u16x4*)xl)[t] = l;
}

// --------------------------- kernel 4: WMMA GEMM ---------------------------
// Block: 256 threads = 8 waves (2 M x 4 N). Wave tile 64x32 (4x2 WMMA tiles).
// Block tile 128(M) x 128(N).  K = 2048, stepped by 32.
__global__ __launch_bounds__(256) void k_gemm(const unsigned short* __restrict__ xh,
                                              const unsigned short* __restrict__ xl,
                                              const unsigned short* __restrict__ wh,
                                              const unsigned short* __restrict__ wl,
                                              const float* __restrict__ bias,
                                              float* __restrict__ out) {
    const int lane  = threadIdx.x & 31;
    const int wave  = threadIdx.x >> 5;
    const int wm    = wave >> 2;                  // 0..1
    const int wn    = wave & 3;                   // 0..3
    const int mbase = blockIdx.y * 128 + wm * 64;
    const int nbase = blockIdx.x * 128 + wn * 32;
    const int lhalf = lane >> 4;                  // 0/1
    const int lmod  = lane & 15;

    // A frag (16x32 bf16): lane holds row M=lmod; 8-K chunks at (lhalf*8, +16)
    const int aoff = lhalf * 8;
    // B frag (32x16 bf16): lane holds col N=lmod; 8-K chunks at (lhalf*8, +16)
    const int boff = lhalf * 8;

    int arow[4], brow[2];
    #pragma unroll
    for (int mi = 0; mi < 4; ++mi)
        arow[mi] = (mbase + mi * 16 + lmod) * INPUTDIM;
    #pragma unroll
    for (int ni = 0; ni < 2; ++ni)
        brow[ni] = (nbase + ni * 16 + lmod) * INPUTDIM;

    const v8f zero = {0.f, 0.f, 0.f, 0.f, 0.f, 0.f, 0.f, 0.f};
    v8f acc[4][2] = {{zero, zero}, {zero, zero}, {zero, zero}, {zero, zero}};

    for (int k0 = 0; k0 < INPUTDIM; k0 += 32) {
        FragU ah[4], al[4], bh[2], bl[2];
        #pragma unroll
        for (int ni = 0; ni < 2; ++ni) {
            const unsigned short* ph = wh + brow[ni] + k0 + boff;
            bh[ni].q[0] = *(const u32x4*)(ph);
            bh[ni].q[1] = *(const u32x4*)(ph + 16);
            const unsigned short* pl = wl + brow[ni] + k0 + boff;
            bl[ni].q[0] = *(const u32x4*)(pl);
            bl[ni].q[1] = *(const u32x4*)(pl + 16);
        }
        #pragma unroll
        for (int mi = 0; mi < 4; ++mi) {
            const unsigned short* ph = xh + arow[mi] + k0 + aoff;
            ah[mi].q[0] = *(const u32x4*)(ph);
            ah[mi].q[1] = *(const u32x4*)(ph + 16);
            const unsigned short* pl = xl + arow[mi] + k0 + aoff;
            al[mi].q[0] = *(const u32x4*)(pl);
            al[mi].q[1] = *(const u32x4*)(pl + 16);
        }
        #pragma unroll
        for (int mi = 0; mi < 4; ++mi) {
            #pragma unroll
            for (int ni = 0; ni < 2; ++ni) {
                acc[mi][ni] = __builtin_amdgcn_wmma_f32_16x16x32_bf16(
                    false, ah[mi].v, false, bh[ni].v, (short)0, acc[mi][ni], false, false);
                acc[mi][ni] = __builtin_amdgcn_wmma_f32_16x16x32_bf16(
                    false, ah[mi].v, false, bl[ni].v, (short)0, acc[mi][ni], false, false);
                acc[mi][ni] = __builtin_amdgcn_wmma_f32_16x16x32_bf16(
                    false, al[mi].v, false, bh[ni].v, (short)0, acc[mi][ni], false, false);
            }
        }
    }

    // Epilogue: C/D layout -> row = r + lhalf*8, col = lmod
    #pragma unroll
    for (int ni = 0; ni < 2; ++ni) {
        int col = nbase + ni * 16 + lmod;
        float bv = bias[col];
        #pragma unroll
        for (int mi = 0; mi < 4; ++mi) {
            #pragma unroll
            for (int r = 0; r < 8; ++r) {
                int row = mbase + mi * 16 + lhalf * 8 + r;
                out[row * OUTDIM + col] = acc[mi][ni][r] + bv;
            }
        }
    }
}

// ---------------------------------------------------------------------------
extern "C" void kernel_launch(void* const* d_in, const int* in_sizes, int n_in,
                              void* d_out, int out_size, void* d_ws, size_t ws_size,
                              hipStream_t stream) {
    const float* x    = (const float*)d_in[0];   // [4096, 2048]
    const float* fc   = (const float*)d_in[1];   // [2, 1024, 15, 64]
    const float* bias = (const float*)d_in[2];   // [1, 1024]
    float* out = (float*)d_out;                  // [4096, 1024]

    // Workspace layout (bytes):
    //   [0,       64K)   cwT
    //   [64K,    128K)   swT
    //   [128K,   128K+4M)      wh  (1024*2048 bf16)
    //   [+4M,    +8M)          wl
    //   [+8M,    +24M)         xh  (4096*2048 bf16)
    //   [+24M,   +40M)         xl
    char* ws = (char*)d_ws;
    float* cwT = (float*)(ws);
    float* swT = (float*)(ws + (1 << 16));
    unsigned short* wh = (unsigned short*)(ws + (2 << 16));
    unsigned short* wl = wh + (size_t)OUTDIM * INPUTDIM;
    unsigned short* xh = (unsigned short*)(ws + (2 << 16) +
                                           2 * (size_t)OUTDIM * INPUTDIM * sizeof(unsigned short));
    unsigned short* xl = xh + (size_t)BATCH * INPUTDIM;

    k_tables<<<64, 256, 0, stream>>>(cwT, swT);
    k_buildw<<<(OUTDIM * INPUTDIM) / 256, 256, 0, stream>>>(fc, cwT, swT, wh, wl);
    k_splitx<<<(BATCH * INPUTDIM / 4) / 256, 256, 0, stream>>>(x, xh, xl);
    k_gemm<<<dim3(OUTDIM / 128, BATCH / 128), 256, 0, stream>>>(xh, xl, wh, wl, bias, out);
}
